// NeighborhoodFeatureSampling_90202903151301
// MI455X (gfx1250) — matrix operand order repaired and checked by reading, not compiled, not run on hardware
//
#include <hip/hip_runtime.h>
#include <hip/hip_bf16.h>

typedef _Float16 v16h __attribute__((ext_vector_type(16)));
typedef _Float16 v8h  __attribute__((ext_vector_type(8)));
typedef float    v8f  __attribute__((ext_vector_type(8)));

// Problem constants (match reference)
#define BATCH 2
#define CCH   32
#define HH    96
#define WW    96
#define DD    96
#define NV    30000
#define HWD   (HH * WW * DD)
#define KNB   27           // 27 neighbor offsets
#define KDIM  (KNB * CCH)  // 864 reduction depth
#define VPB   16           // vertices per workgroup (one WMMA tile)

// ---------------------------------------------------------------------------
// Kernel 1: pack conv_w (C_out=32, C_in=32, 1, 27) f32 -> f16 B-matrix
// fragments in the exact per-lane VGPR layout of V_WMMA_F32_16X16X32_F16.
//   flat index = ((kk*2 + h)*32 + lane)*16 + e
//   lane: column o = (lane&15) + h*16 ; K(c_in) = (lane<16 ? 0 : 8) + (e<8 ? e : e+8)
// ---------------------------------------------------------------------------
__global__ void pack_weights_f16(const float* __restrict__ w,
                                 _Float16* __restrict__ bp) {
    int t = blockIdx.x * blockDim.x + threadIdx.x;
    if (t >= KNB * 2 * 32 * 16) return;
    int e    = t & 15;
    int lane = (t >> 4) & 31;
    int h    = (t >> 9) & 1;
    int kk   = t >> 10;
    int o  = (lane & 15) + h * 16;
    int ci = ((lane < 16) ? 0 : 8) + ((e < 8) ? e : e + 8);
    bp[t] = (_Float16)w[(o * CCH + ci) * KNB + kk];
}

// Build per-axis 4-wide interpolation weights for the 3 offsets {-1,0,+1}.
// Works uniformly for interior AND clamped (border) positions:
//   pos = clip(pbase + o - 1); ii0 = floor(pos); f = frac
//   j0  = ii0 - (base0-1) is always in {0,1,2}, and the clamped corner block
//   cc[j] = clamp(base0-1+j) satisfies cc[j0]==ii0, cc[j0+1]==ii1.
__device__ __forceinline__ void build_axis_w(float pb, int i0m1, int dmax,
                                             float (&Wm)[3][4]) {
    #pragma unroll
    for (int o = 0; o < 3; ++o) {
        float pos = fminf(fmaxf(pb + (float)(o - 1), 0.f), (float)dmax);
        int   ii0 = (int)pos;
        float ff  = pos - (float)ii0;
        int   j0  = ii0 - i0m1;
        #pragma unroll
        for (int j = 0; j < 4; ++j)
            Wm[o][j] = (j == j0) ? (1.f - ff) : ((j == j0 + 1) ? ff : 0.f);
    }
}

// ---------------------------------------------------------------------------
// Kernel 2 (fused): shared-corner separable trilinear sampling -> LDS (f16)
// -> WMMA GEMM.  Block = 64 threads (2 waves), 16 vertices (one tile).
// Phase 1: thread = (vertex, channel-quarter). Per channel: load the 4x4x4
//          shared corner block ONCE (64 gathers instead of 27*8=216) and
//          produce all 27 neighbor samples by separable x/y/z reduction.
// Phase 2: wave 0 runs the 16x32 = A(16x864) x B(864x32) WMMA loop.
// ---------------------------------------------------------------------------
__global__ __launch_bounds__(64)
void fused_sample_gemm(const float* __restrict__ vox,
                       const float* __restrict__ verts,
                       const _Float16* __restrict__ bp,
                       const float* __restrict__ bias,
                       float* __restrict__ out) {
    __shared__ __align__(16) _Float16 sA[VPB * KDIM]; // 16*864*2 = 27648 B

    const int b   = blockIdx.y;
    const int v0  = blockIdx.x * VPB;   // N = 1875*16 exactly: no guards
    const int tid = threadIdx.x;

    // ---------------- Phase 1: gather + separable trilinear ----------------
    {
        const int vloc = tid >> 2;          // 0..15
        const int c0   = (tid & 3) * 8;     // channel quarter: 8 channels
        const int vg   = v0 + vloc;

        const float* vp = verts + ((size_t)b * NV + vg) * 3;
        float x = vp[0], y = vp[1], z = vp[2];
        float pbx = (x + 1.f) * 0.5f * (DD - 1);   // x -> D axis
        float pby = (y + 1.f) * 0.5f * (WW - 1);   // y -> W axis
        float pbz = (z + 1.f) * 0.5f * (HH - 1);   // z -> H axis
        int x0 = (int)pbx, y0 = (int)pby, z0 = (int)pbz;

        float Wx[3][4], Wy[3][4], Wz[3][4];
        build_axis_w(pbx, x0 - 1, DD - 1, Wx);
        build_axis_w(pby, y0 - 1, WW - 1, Wy);
        build_axis_w(pbz, z0 - 1, HH - 1, Wz);

        int cx[4], cy4[4], cz4[4];
        #pragma unroll
        for (int j = 0; j < 4; ++j) {
            cx[j]  = min(max(x0 - 1 + j, 0), DD - 1);
            cy4[j] = min(max(y0 - 1 + j, 0), WW - 1);
            cz4[j] = min(max(z0 - 1 + j, 0), HH - 1);
        }
        int rowoff[16];
        #pragma unroll
        for (int zc = 0; zc < 4; ++zc)
            #pragma unroll
            for (int yc = 0; yc < 4; ++yc)
                rowoff[zc * 4 + yc] = (cz4[zc] * WW + cy4[yc]) * DD;

        const float* p = vox + (size_t)b * CCH * HWD + (size_t)c0 * HWD;
        _Float16* dstv = &sA[vloc * KDIM];

        for (int ci = 0; ci < 8; ++ci) {
            float acc[3][3][3] = {};            // [ox][oy][oz]
            #pragma unroll
            for (int zc = 0; zc < 4; ++zc) {
                float ry[3][3] = {};            // [oy][ox]
                #pragma unroll
                for (int yc = 0; yc < 4; ++yc) {
                    const float* row = p + rowoff[zc * 4 + yc];
                    float va = row[cx[0]], vb = row[cx[1]];
                    float vc = row[cx[2]], vd = row[cx[3]];
                    #pragma unroll
                    for (int ox = 0; ox < 3; ++ox) {
                        float rx = va * Wx[ox][0] + vb * Wx[ox][1]
                                 + vc * Wx[ox][2] + vd * Wx[ox][3];
                        #pragma unroll
                        for (int oy = 0; oy < 3; ++oy)
                            ry[oy][ox] += rx * Wy[oy][yc];
                    }
                }
                #pragma unroll
                for (int oz = 0; oz < 3; ++oz) {
                    float wz = Wz[oz][zc];
                    #pragma unroll
                    for (int oy = 0; oy < 3; ++oy)
                        #pragma unroll
                        for (int ox = 0; ox < 3; ++ox)
                            acc[ox][oy][oz] += ry[oy][ox] * wz;
                }
            }
            // k = ox*9 + oy*3 + oz (itertools.product order), kappa = k*32 + c
            #pragma unroll
            for (int ox = 0; ox < 3; ++ox)
                #pragma unroll
                for (int oy = 0; oy < 3; ++oy)
                    #pragma unroll
                    for (int oz = 0; oz < 3; ++oz)
                        dstv[(ox * 9 + oy * 3 + oz) * CCH + c0 + ci] =
                            (_Float16)acc[ox][oy][oz];
            p += HWD;
        }
    }
    __syncthreads();

    // ---------------- Phase 2: WMMA GEMM (wave 0, one 16-vertex tile) ------
    if (tid < 32) {                       // wave-uniform: EXEC all-ones
        const int lane = tid;
        v8f acc0 = {};
        v8f acc1 = {};
        const int m   = lane & 15;                  // A-matrix row
        const int klo = (lane < 16) ? 0 : 8;        // A K sub-offset per lane
        const _Float16* arow = &sA[m * KDIM + klo];
        #pragma unroll 3
        for (int kk = 0; kk < KNB; ++kk) {
            // A: lanes 0-15 hold K {0..7,16..23}, lanes 16-31 {8..15,24..31}
            v8h lo = *(const v8h*)(arow + kk * 32);
            v8h hi = *(const v8h*)(arow + kk * 32 + 16);
            v16h a = __builtin_shufflevector(lo, hi,
                        0, 1, 2, 3, 4, 5, 6, 7, 8, 9, 10, 11, 12, 13, 14, 15);
            v16h b0 = *(const v16h*)(bp + ((kk * 2 + 0) * 32 + lane) * 16);
            v16h b1 = *(const v16h*)(bp + ((kk * 2 + 1) * 32 + lane) * 16);
            acc0 = __builtin_amdgcn_wmma_f32_16x16x32_f16(
                       false, a, false, b0, (short)0, acc0, false, false);
            acc1 = __builtin_amdgcn_wmma_f32_16x16x32_f16(
                       false, a, false, b1, (short)0, acc1, false, false);
        }
        // D layout: VGPR r -> M = r + (lane<16 ? 0 : 8), N = lane&15
        const int ocol  = lane & 15;
        const float bs0 = bias[ocol];
        const float bs1 = bias[ocol + 16];
        const int mbase = (lane < 16) ? 0 : 8;
        #pragma unroll
        for (int r = 0; r < 8; ++r) {
            int vg = v0 + mbase + r;
            float* op = out + ((size_t)b * NV + vg) * CCH;
            op[ocol]      = acc0[r] + bs0;
            op[ocol + 16] = acc1[r] + bs1;
        }
    }
}

extern "C" void kernel_launch(void* const* d_in, const int* in_sizes, int n_in,
                              void* d_out, int out_size, void* d_ws, size_t ws_size,
                              hipStream_t stream) {
    const float* voxel_features = (const float*)d_in[0];
    const float* vertices       = (const float*)d_in[1];
    const float* conv_w         = (const float*)d_in[2];
    const float* conv_b         = (const float*)d_in[3];
    // d_in[4..6] = img_H/img_W/img_D scalars; compile-time constants used.
    float* out = (float*)d_out;

    _Float16* bpack = (_Float16*)d_ws;   // 27*2*32*16 halfs = 55,296 B

    // 1) Pack weights into WMMA B-fragment layout (one-shot, tiny).
    pack_weights_f16<<<dim3((KNB * 1024) / 256), 256, 0, stream>>>(conv_w, bpack);

    // 2) Fused sampling + GEMM: one 16-vertex tile per block, N = 1875*16.
    fused_sample_gemm<<<dim3(NV / VPB, BATCH), 64, 0, stream>>>(
        voxel_features, vertices, bpack, conv_b, out);
}
